// SpatialRVQ_8976481648790
// MI455X (gfx1250) — matrix operand order repaired and checked by reading, not compiled, not run on hardware
//
#include <hip/hip_runtime.h>

typedef __attribute__((ext_vector_type(16))) _Float16 v16h;
typedef __attribute__((ext_vector_type(8)))  float    v8f;
typedef int v4i __attribute__((vector_size(16)));

#define R_STAGES 6
#define KCODES   512
#define CDIM     64
#define TOKPB    128     // tokens per block
#define THREADS  256     // 8 wave32
#define PITCH    66      // f32 LDS pitch for 64 channels (bank-conflict pad)

#define AS1 __attribute__((address_space(1)))
#define AS3 __attribute__((address_space(3)))

#if defined(__gfx1250__) && \
    __has_builtin(__builtin_amdgcn_global_load_async_to_lds_b128) && \
    __has_builtin(__builtin_amdgcn_s_wait_asynccnt)
#define USE_ASYNC_LDS 1
#else
#define USE_ASYNC_LDS 0
#endif

// ---------------- prep: f32 codebook -> f16 + squared norms ----------------
__global__ void rvq_prep(const float* __restrict__ embed,
                         _Float16* __restrict__ cbh,
                         float* __restrict__ norms) {
    int code = blockIdx.x * blockDim.x + threadIdx.x;   // 0..R*K-1
    if (code >= R_STAGES * KCODES) return;
    const float* e = embed + (size_t)code * CDIM;
    float s = 0.f;
#pragma unroll
    for (int c = 0; c < CDIM; ++c) {
        float v = e[c];
        s += v * v;
        cbh[(size_t)code * CDIM + c] = (_Float16)v;
    }
    norms[code] = s;
}

// ---------------- zero usage flags + loss accumulator ----------------
__global__ void rvq_zero(unsigned* __restrict__ flags, float* __restrict__ loss) {
    int i = blockIdx.x * blockDim.x + threadIdx.x;
    if (i < R_STAGES * KCODES) flags[i] = 0u;
    if (i == R_STAGES * KCODES) *loss = 0.f;
}

// ---------------- main: residual VQ over 6 stages, WMMA distances ----------------
__global__ __launch_bounds__(THREADS)
void rvq_main(const float* __restrict__ zin,
              const float* __restrict__ embed,      // f32 codebooks (exact residual update)
              const _Float16* __restrict__ cbh,     // f16 codebooks [R][K][C]
              const float* __restrict__ norms,      // ||e||^2 [R][K]
              unsigned* __restrict__ flags,
              float* __restrict__ loss_acc,
              float* __restrict__ zq_out,
              float* __restrict__ codes_out) {
    __shared__ float    s_resid[TOKPB * PITCH];   // 33792 B, f32 residual
    __shared__ _Float16 s_cb[KCODES * CDIM];      // 65536 B, stage codebook f16
    __shared__ float    s_norm[KCODES];           // 2048 B
    __shared__ float    s_red[THREADS];           // 1024 B

    const int tid  = threadIdx.x;
    const int lane = tid & 31;
    const int wave = tid >> 5;
    const int tokenBase = blockIdx.x * TOKPB;
    const int b  = tokenBase >> 12;      // 4096 tokens per image
    const int sp = tokenBase & 4095;     // contiguous spatial offset (h*64+w)

    // Load z: [B,C,H,W] layout; tokens contiguous along spatial -> coalesced per channel.
    for (int j = tid; j < TOKPB * CDIM; j += THREADS) {
        int c = j >> 7, i = j & 127;
        s_resid[i * PITCH + c] = zin[(size_t)(b * CDIM + c) * 4096 + sp + i];
    }
    __syncthreads();

    const int m    = lane & 15;          // output row within wave tile / B column
    const int kh   = lane >> 4;          // lane half selects K sub-range
    const int trow = (wave * 16 + m) * PITCH;

    for (int r = 0; r < R_STAGES; ++r) {
        // ---- stage codebook (f16) + norms into LDS once per block per stage ----
#if USE_ASYNC_LDS
        {
            const uint4* src = (const uint4*)(cbh + (size_t)r * KCODES * CDIM);
            uint4* dst = (uint4*)s_cb;
            for (int j = tid; j < KCODES * CDIM / 8; j += THREADS) {
                __builtin_amdgcn_global_load_async_to_lds_b128(
                    (AS1 v4i*)(uintptr_t)(src + j),
                    (AS3 v4i*)(unsigned)(uintptr_t)(dst + j),
                    0, 0);
            }
            for (int j = tid; j < KCODES; j += THREADS) s_norm[j] = norms[r * KCODES + j];
            __builtin_amdgcn_s_wait_asynccnt(0);
        }
#else
        {
            const uint4* src = (const uint4*)(cbh + (size_t)r * KCODES * CDIM);
            uint4* dst = (uint4*)s_cb;
            for (int j = tid; j < KCODES * CDIM / 8; j += THREADS) dst[j] = src[j];
            for (int j = tid; j < KCODES; j += THREADS) s_norm[j] = norms[r * KCODES + j];
        }
#endif
        __syncthreads();

        // Prefetch next stage's codebook into L2 while this stage computes.
        if (r + 1 < R_STAGES) {
            const char* nxt = (const char*)(cbh + (size_t)(r + 1) * KCODES * CDIM);
            __builtin_prefetch(nxt + tid * 256, 0, 0);
            __builtin_prefetch(nxt + tid * 256 + 128, 0, 0);
        }

        // A fragments: lane half kh covers contiguous K {kh*16..+15} (matches B fragment loads).
        v16h a0, a1;
#pragma unroll
        for (int i = 0; i < 16; ++i) {
            int kk = kh * 16 + i;
            a0[i] = (_Float16)s_resid[trow + kk];        // K = 0..31
            a1[i] = (_Float16)s_resid[trow + 32 + kk];   // K = 32..63
        }

        float bestd[8];
        int   bestn[8];
#pragma unroll
        for (int v = 0; v < 8; ++v) { bestd[v] = 3.0e38f; bestn[v] = 0; }

        // Sweep all 512 codes in 16-wide tiles, software-pipelined one tile ahead.
        // dist = ||e||^2 - 2 r.e  (||r||^2 is row-constant, dropped under argmin)
        const int roff = m * CDIM + kh * 16;             // halves
        v16h b0 = *(const v16h*)(s_cb + roff);
        v16h b1 = *(const v16h*)(s_cb + roff + 32);
        float nn = s_norm[m];
        for (int nt = 0; nt < 32; ++nt) {
            // prefetch next tile's fragments (wraps to tile 0 on last iter; harmless)
            const int noff = roff + (((nt + 1) & 31) << 10);
            v16h p0 = *(const v16h*)(s_cb + noff);
            v16h p1 = *(const v16h*)(s_cb + noff + 32);
            float pn = s_norm[(((nt + 1) & 31) << 4) + m];

            v8f acc = {};
            acc = __builtin_amdgcn_wmma_f32_16x16x32_f16(false, a0, false, b0,
                                                         (short)0, acc, false, false);
            acc = __builtin_amdgcn_wmma_f32_16x16x32_f16(false, a1, false, b1,
                                                         (short)0, acc, false, false);
            const int n = (nt << 4) + m;
#pragma unroll
            for (int v = 0; v < 8; ++v) {
                float d = nn - 2.0f * acc[v];
                if (d < bestd[v]) { bestd[v] = d; bestn[v] = n; }
            }
            b0 = p0; b1 = p1; nn = pn;
        }

        // Branchless butterfly argmin across the 16 lanes holding each row
        // (index tie-break = first, matching jnp.argmin).
#pragma unroll
        for (int mask = 1; mask <= 8; mask <<= 1) {
#pragma unroll
            for (int v = 0; v < 8; ++v) {
                float od = __shfl_xor(bestd[v], mask, 32);
                int   on = __shfl_xor(bestn[v], mask, 32);
                int take = (int)(od < bestd[v]) |
                           ((int)(od == bestd[v]) & (int)(on < bestn[v]));
                bestd[v] = take ? od : bestd[v];
                bestn[v] = take ? on : bestn[v];
            }
        }

        // Swap halves so every lane knows all 16 winners of its wave tile.
        int idxAll[16];
#pragma unroll
        for (int v = 0; v < 8; ++v) {
            int on = __shfl_xor(bestn[v], 16, 32);
            int a = (kh == 0) ? bestn[v] : on;
            int c = (kh == 0) ? on : bestn[v];
            idxAll[v]     = a;
            idxAll[v + 8] = c;
        }

        // One lane per half writes codes + usage flags (rows v + 8*kh).
        if (m == 0) {
#pragma unroll
            for (int v = 0; v < 8; ++v) {
                int t = tokenBase + wave * 16 + v + 8 * kh;
                codes_out[(size_t)t * R_STAGES + r] = (float)bestn[v];
                flags[r * KCODES + bestn[v]] = 1u;       // benign race: all write 1
            }
        }

        // Exact f32 residual update from original codebook (lane covers 2 channels).
        for (int t = 0; t < 16; ++t) {
            const float2* e =
                (const float2*)(embed + ((size_t)r * KCODES + idxAll[t]) * CDIM);
            float2 ev = e[lane];
            float* rr = &s_resid[(wave * 16 + t) * PITCH + 2 * lane];
            rr[0] -= ev.x;
            rr[1] -= ev.y;
        }
        __syncthreads();   // protect s_cb reuse next stage
    }

    // z_q = z - final residual; accumulate ||final residual||^2 for recon loss.
    float lsum = 0.f;
    for (int j = tid; j < TOKPB * CDIM; j += THREADS) {
        int c = j >> 7, i = j & 127;
        float rr = s_resid[i * PITCH + c];
        size_t g = (size_t)(b * CDIM + c) * 4096 + sp + i;
        zq_out[g] = zin[g] - rr;
        lsum += rr * rr;
    }
    s_red[tid] = lsum;
    __syncthreads();
    for (int s = THREADS / 2; s > 0; s >>= 1) {
        if (tid < s) s_red[tid] += s_red[tid + s];
        __syncthreads();
    }
    if (tid == 0) atomicAdd(loss_acc, s_red[0]);
}

// ---------------- finalize: loss mean + per-stage usage ----------------
__global__ void rvq_finalize(const unsigned* __restrict__ flags,
                             const float* __restrict__ loss,
                             float* __restrict__ loss_out,
                             float* __restrict__ usage_out) {
    int t = threadIdx.x;
    if (t < R_STAGES) {
        int cnt = 0;
        for (int k = 0; k < KCODES; ++k) cnt += (flags[t * KCODES + k] != 0u);
        usage_out[t] = (float)cnt * (1.0f / KCODES);
    }
    if (t == 32) loss_out[0] = loss[0] * (1.0f / 4194304.0f);  // mean over B*C*H*W
}

// ---------------- launch ----------------
extern "C" void kernel_launch(void* const* d_in, const int* in_sizes, int n_in,
                              void* d_out, int out_size, void* d_ws, size_t ws_size,
                              hipStream_t stream) {
    const float* zin   = (const float*)d_in[0];   // [16,64,64,64]
    const float* embed = (const float*)d_in[1];   // [6,512,64]

    char* ws = (char*)d_ws;
    _Float16* cbh   = (_Float16*)ws;               // 393216 B f16 codebooks
    float*    norms = (float*)(ws + 393216);       // 12288 B
    unsigned* flags = (unsigned*)(ws + 405504);    // 12288 B
    float*    loss  = (float*)(ws + 417792);       // 4 B

    float* out       = (float*)d_out;
    float* zq_out    = out;                        // 4194304 floats [B,C,H,W]
    float* codes_out = out + 4194304;              // 393216 floats  [B,H,W,R]
    float* loss_out  = out + 4587520;              // 1 float
    float* usage_out = out + 4587521;              // 6 floats

    rvq_prep<<<(R_STAGES * KCODES + 255) / 256, 256, 0, stream>>>(embed, cbh, norms);
    rvq_zero<<<(R_STAGES * KCODES + 256) / 256 + 1, 256, 0, stream>>>(flags, loss);
    rvq_main<<<65536 / TOKPB, THREADS, 0, stream>>>(zin, embed, cbh, norms,
                                                    flags, loss, zq_out, codes_out);
    rvq_finalize<<<1, 64, 0, stream>>>(flags, loss, loss_out, usage_out);
}
